// T5Attention_85031762526250
// MI455X (gfx1250) — matrix operand order repaired and checked by reading, not compiled, not run on hardware
//
#include <hip/hip_runtime.h>
#include <hip/hip_bf16.h>

// Problem constants (match reference)
#define B_  8
#define S_  512
#define KV_ 2048
#define NH_ 16
#define HD_ 64
#define D_  (NH_ * HD_)   // 1024
#define M_  (B_ * S_)     // 4096

typedef __attribute__((ext_vector_type(16))) _Float16 v16h_t;
typedef __attribute__((ext_vector_type(8)))  _Float16 v8h_t;
typedef __attribute__((ext_vector_type(8)))  float    v8f_t;

// ---------------------------------------------------------------------------
// WMMA fragment loaders (wave32), contiguous k order shared by A and B
// A 16x32 f16 (MxK): M=lane&15; elem j: k=half*16+j
// ---------------------------------------------------------------------------
__device__ __forceinline__ v16h_t load_frag_a(const _Float16* __restrict__ src,
                                              int ld, int lane) {
  const int m    = lane & 15;
  const int half = lane >> 4;
  const _Float16* row = src + m * ld;
  v16h_t a = *(const v16h_t*)(row + half * 16);      // k = half*16 + 0..15
  return a;
}

// B 32x16 f16 (KxN), source laid out column-contiguous-in-K:
// element j: n = lane&15, k = (lane>>4)*16 + j  -> one 32B contiguous read
__device__ __forceinline__ v16h_t load_frag_b(const _Float16* __restrict__ src,
                                              int ld, int lane) {
  const int n    = lane & 15;
  const int half = lane >> 4;
  return *(const v16h_t*)(src + n * ld + half * 16);
}

__device__ __forceinline__ v8f_t wmma_f16(v16h_t a, v16h_t b, v8f_t c) {
  return __builtin_amdgcn_wmma_f32_16x16x32_f16(false, a, false, b,
                                                (short)0, c, false, false);
}

// ---------------------------------------------------------------------------
// Elementwise f32 -> f16 convert (grid-stride)
// ---------------------------------------------------------------------------
__global__ void __launch_bounds__(256) cvt_f16_kernel(const float* __restrict__ src,
                                                      _Float16* __restrict__ dst,
                                                      int n) {
  for (int i = blockIdx.x * blockDim.x + threadIdx.x; i < n;
       i += gridDim.x * blockDim.x)
    dst[i] = (_Float16)src[i];
}

// past_value [b,h,kv,d] f32 -> vt [b,h,d,kv] f16  (coalesced reads)
__global__ void __launch_bounds__(256) vtrans_kernel(const float* __restrict__ v,
                                                     _Float16* __restrict__ vt) {
  int idx = blockIdx.x * blockDim.x + threadIdx.x;      // over B*NH*KV*HD = 2^24
  int d  = idx & (HD_ - 1);
  int n  = (idx >> 6) & (KV_ - 1);
  int bh = idx >> 17;
  vt[(bh * HD_ + d) * KV_ + n] = (_Float16)v[idx];
}

// ---------------------------------------------------------------------------
// Q projection: q[m,n] = sum_k hid[m,k] * Wq[n,k]   (f16 in, f32 acc)
// block = 128 (4 waves); wave -> 16 rows x 64 cols.
// K-loop unrolled x2 with two ping-pong fragment sets: set A computes while
// set B's loads are in flight, then roles swap -> no register-copy churn.
// ---------------------------------------------------------------------------
__global__ void __launch_bounds__(128) qproj_kernel(const _Float16* __restrict__ hid,
                                                    const _Float16* __restrict__ wq,
                                                    _Float16* __restrict__ qout) {
  const int lane = threadIdx.x & 31;
  const int wave = threadIdx.x >> 5;
  const int nGroups = D_ / 64;                       // 16
  const int blockM  = blockIdx.x / nGroups;          // 0..63
  const int blockN  = blockIdx.x % nGroups;          // 0..15
  const int mbase   = blockM * 64 + wave * 16;
  const int nbase0  = blockN * 64;

  const _Float16* ha = hid + mbase * D_;
  const _Float16* wb = wq + nbase0 * D_;

  v8f_t acc[4];
#pragma unroll
  for (int t = 0; t < 4; ++t) acc[t] = (v8f_t)0.0f;

  // ---- prologue: set A <- k = 0 ----
  v16h_t aA = load_frag_a(ha, D_, lane);
  v16h_t bA[4];
#pragma unroll
  for (int nt = 0; nt < 4; ++nt) bA[nt] = load_frag_b(wb + nt * 16 * D_, D_, lane);

  v16h_t aB, bB[4];
  for (int k0 = 0; k0 < D_; k0 += 64) {
    {   // stage 1: fill set B (k0+32), compute set A (k0)
      const int kn = k0 + 32;                        // <= 992, in range
      aB = load_frag_a(ha + kn, D_, lane);
#pragma unroll
      for (int nt = 0; nt < 4; ++nt)
        bB[nt] = load_frag_b(wb + nt * 16 * D_ + kn, D_, lane);
#pragma unroll
      for (int nt = 0; nt < 4; ++nt) acc[nt] = wmma_f16(aA, bA[nt], acc[nt]);
    }
    {   // stage 2: fill set A (k0+64, wraps), compute set B (k0+32)
      const int kn = (k0 + 64) & (D_ - 1);
      aA = load_frag_a(ha + kn, D_, lane);
#pragma unroll
      for (int nt = 0; nt < 4; ++nt)
        bA[nt] = load_frag_b(wb + nt * 16 * D_ + kn, D_, lane);
#pragma unroll
      for (int nt = 0; nt < 4; ++nt) acc[nt] = wmma_f16(aB, bB[nt], acc[nt]);
    }
  }

  const int half = lane >> 4, ln = lane & 15;
#pragma unroll
  for (int nt = 0; nt < 4; ++nt) {
    const int n = nbase0 + nt * 16 + ln;
    const int h = n >> 6, d = n & (HD_ - 1);
#pragma unroll
    for (int r = 0; r < 8; ++r) {
      const int m = mbase + r + 8 * half;
      const int b = m >> 9, s = m & (S_ - 1);
      qout[(((b * NH_ + h) * S_ + s) << 6) + d] = (_Float16)acc[nt][r];
    }
  }
}

// ---------------------------------------------------------------------------
// One 32-wide KV tile of flash attention for one wave (16 query rows).
// Consumes K fragments `bkc` (this tile), fills `bkn` with the next tile's
// fragments (loads issued before the softmax VALU section hides them).
// ---------------------------------------------------------------------------
__device__ __forceinline__ void attn_tile(int kv, int lane, int half, int ln,
                                          const _Float16* __restrict__ kp,
                                          const _Float16* __restrict__ vp,
                                          const float* __restrict__ mp,
                                          _Float16* myp,
                                          const v16h_t& aq0, const v16h_t& aq1,
                                          v16h_t (&bkc)[4], v16h_t (&bkn)[4],
                                          v8f_t (&o)[4],
                                          float (&mrow)[8], float (&lrow)[8]) {
  // V fragments for this tile: issue loads NOW, consume after softmax
  v16h_t bv[4];
#pragma unroll
  for (int t = 0; t < 4; ++t)
    bv[t] = load_frag_b(vp + (t * 16) * KV_ + kv, KV_, lane);

  // prefetch K/V tiles 4 iterations ahead into caches (global_prefetch_b8)
  {
    const int kvp = (kv + 128) & (KV_ - 1);
    __builtin_prefetch(kp + kvp * HD_ + lane * 64, 0, 1);   // 32 lanes x 128B = 4KB K tile
    __builtin_prefetch(vp + (size_t)(lane * 2) * KV_ + kvp, 0, 1);
    __builtin_prefetch(vp + (size_t)(lane * 2 + 1) * KV_ + kvp, 0, 1);
  }

  // scores S = Q . K^T  (16 x 32, two 16x16 D fragments)
  v8f_t sf0 = (v8f_t)0.0f, sf1 = (v8f_t)0.0f;
  sf0 = wmma_f16(aq0, bkc[0], sf0);
  sf0 = wmma_f16(aq1, bkc[1], sf0);
  sf1 = wmma_f16(aq0, bkc[2], sf1);
  sf1 = wmma_f16(aq1, bkc[3], sf1);

  // issue next K-tile fragment loads; they complete under softmax
  const int kvn = (kv + 32) & (KV_ - 1);
  bkn[0] = load_frag_b(kp + (kvn +  0) * HD_ +  0, HD_, lane);
  bkn[1] = load_frag_b(kp + (kvn +  0) * HD_ + 32, HD_, lane);
  bkn[2] = load_frag_b(kp + (kvn + 16) * HD_ +  0, HD_, lane);
  bkn[3] = load_frag_b(kp + (kvn + 16) * HD_ + 32, HD_, lane);

  const float mv0 = mp[kv + ln];
  const float mv1 = mp[kv + 16 + ln];

  // online softmax per row (row = r + 8*half; 16 lanes per row)
#pragma unroll
  for (int r = 0; r < 8; ++r) {
    float e0 = sf0[r] + mv0;
    float e1 = sf1[r] + mv1;
    float mx = fmaxf(e0, e1);
    mx = fmaxf(mx, __shfl_xor(mx, 1, 32));
    mx = fmaxf(mx, __shfl_xor(mx, 2, 32));
    mx = fmaxf(mx, __shfl_xor(mx, 4, 32));
    mx = fmaxf(mx, __shfl_xor(mx, 8, 32));
    const float nm    = fmaxf(mrow[r], mx);
    const float scale = __expf(mrow[r] - nm);
    const float p0 = __expf(e0 - nm);
    const float p1 = __expf(e1 - nm);
    float rs = p0 + p1;
    rs += __shfl_xor(rs, 1, 32);
    rs += __shfl_xor(rs, 2, 32);
    rs += __shfl_xor(rs, 4, 32);
    rs += __shfl_xor(rs, 8, 32);
    lrow[r] = lrow[r] * scale + rs;
    mrow[r] = nm;
#pragma unroll
    for (int t = 0; t < 4; ++t) o[t][r] *= scale;
    const int m = r + 8 * half;
    myp[m * 32 + ln]      = (_Float16)p0;   // P row-major 16x32 in LDS
    myp[m * 32 + 16 + ln] = (_Float16)p1;
  }
  __builtin_amdgcn_wave_barrier();
  asm volatile("s_wait_dscnt 0" ::: "memory");   // lanes exchanged P via LDS

  // O += P . V  (A = P from LDS, B = V^T tiles, K = 32)
  const v16h_t ap = load_frag_a(myp, 32, lane);
#pragma unroll
  for (int t = 0; t < 4; ++t)
    o[t] = wmma_f16(ap, bv[t], o[t]);
  __builtin_amdgcn_wave_barrier();               // keep LDS buffer reuse ordered
}

// ---------------------------------------------------------------------------
// Flash attention: block = 256 (8 waves) covers 128 queries of one (b,h).
// KV loop unrolled x2 with ping-pong K fragment sets (no register copies).
// ---------------------------------------------------------------------------
__global__ void __launch_bounds__(256) attn_kernel(const _Float16* __restrict__ q,
                                                   const _Float16* __restrict__ kf,
                                                   const _Float16* __restrict__ vt,
                                                   const float* __restrict__ mask,
                                                   float* __restrict__ out) {
  __shared__ __align__(64) _Float16 pbuf[8][16 * 32];   // per-wave P staging

  const int lane = threadIdx.x & 31;
  const int wave = threadIdx.x >> 5;
  const int half = lane >> 4, ln = lane & 15;

  const int bh = blockIdx.x / (S_ / 128);
  const int st = blockIdx.x % (S_ / 128);
  const int b = bh / NH_;
  const int h = bh % NH_;
  const int s0 = st * 128 + wave * 16;

  const _Float16* qp = q  + (bh * S_ + s0) * HD_;
  const _Float16* kp = kf + (size_t)bh * KV_ * HD_;
  const _Float16* vp = vt + (size_t)bh * HD_ * KV_;
  const float*    mp = mask + b * KV_;
  _Float16* myp = pbuf[wave];

  const v16h_t aq0 = load_frag_a(qp + 0,  HD_, lane);   // d = 0..31
  const v16h_t aq1 = load_frag_a(qp + 32, HD_, lane);   // d = 32..63

  v8f_t o[4];
#pragma unroll
  for (int t = 0; t < 4; ++t) o[t] = (v8f_t)0.0f;
  float mrow[8], lrow[8];
#pragma unroll
  for (int r = 0; r < 8; ++r) { mrow[r] = -1.0e30f; lrow[r] = 0.0f; }

  // prologue: K fragments for kv = 0 into set A
  v16h_t bkA[4], bkB[4];
  bkA[0] = load_frag_b(kp +  0 * HD_ +  0, HD_, lane);
  bkA[1] = load_frag_b(kp +  0 * HD_ + 32, HD_, lane);
  bkA[2] = load_frag_b(kp + 16 * HD_ +  0, HD_, lane);
  bkA[3] = load_frag_b(kp + 16 * HD_ + 32, HD_, lane);

  for (int kv = 0; kv < KV_; kv += 64) {
    attn_tile(kv,      lane, half, ln, kp, vp, mp, myp, aq0, aq1,
              bkA, bkB, o, mrow, lrow);
    attn_tile(kv + 32, lane, half, ln, kp, vp, mp, myp, aq0, aq1,
              bkB, bkA, o, mrow, lrow);
  }

  // epilogue: out[b, s, h*64 + d] = o / l
#pragma unroll
  for (int t = 0; t < 4; ++t) {
    const int d = t * 16 + ln;
#pragma unroll
    for (int r = 0; r < 8; ++r) {
      const int s = s0 + r + 8 * half;
      out[((size_t)(b * S_ + s)) * D_ + h * HD_ + d] = o[t][r] / lrow[r];
    }
  }
}

// ---------------------------------------------------------------------------
// Host launch
// ---------------------------------------------------------------------------
extern "C" void kernel_launch(void* const* d_in, const int* in_sizes, int n_in,
                              void* d_out, int out_size, void* d_ws, size_t ws_size,
                              hipStream_t stream) {
  const float* hid  = (const float*)d_in[0];   // [B,S,D]
  const float* wq   = (const float*)d_in[1];   // [D,D]
  const float* pk   = (const float*)d_in[2];   // [B,NH,KV,HD]
  const float* pv   = (const float*)d_in[3];   // [B,NH,KV,HD]
  const float* mask = (const float*)d_in[4];   // [B,KV]
  float* out = (float*)d_out;

  char* ws = (char*)d_ws;
  const size_t nHid = (size_t)M_ * D_;              // 4,194,304
  const size_t nWq  = (size_t)D_ * D_;              // 1,048,576
  const size_t nKV  = (size_t)B_ * NH_ * KV_ * HD_; // 16,777,216
  _Float16* hidh = (_Float16*)(ws);
  _Float16* wqh  = (_Float16*)(ws + nHid * 2);
  _Float16* kh   = (_Float16*)(ws + (nHid + nWq) * 2);
  _Float16* vth  = (_Float16*)(ws + (nHid + nWq + nKV) * 2);
  _Float16* qh   = (_Float16*)(ws + (nHid + nWq + 2 * nKV) * 2);
  // total ws use: (2*nHid + nWq + 2*nKV) * 2 bytes ~= 84 MB

  // 1) precision conversion + V transpose
  cvt_f16_kernel<<<(int)(nHid / 256), 256, 0, stream>>>(hid, hidh, (int)nHid);
  cvt_f16_kernel<<<(int)(nWq  / 256), 256, 0, stream>>>(wq,  wqh,  (int)nWq);
  cvt_f16_kernel<<<(int)(nKV  / 256), 256, 0, stream>>>(pk,  kh,   (int)nKV);
  vtrans_kernel <<<(int)(nKV  / 256), 256, 0, stream>>>(pv,  vth);

  // 2) q projection (WMMA): grid = (M/64) * (D/64) = 64 * 16 blocks of 128
  qproj_kernel<<<(M_ / 64) * (D_ / 64), 128, 0, stream>>>(hidh, wqh, qh);

  // 3) flash attention (WMMA): grid = B*NH*(S/128) = 512 blocks of 256
  attn_kernel<<<B_ * NH_ * (S_ / 128), 256, 0, stream>>>(qh, kh, vth, mask, out);
}
